// Retention_62397284876544
// MI455X (gfx1250) — compile-verified, matrix operand order checked by reading, not run on hardware
//
#include <hip/hip_runtime.h>
#include <hip/hip_bf16.h>
#include <stdint.h>

#define NUM_HEAD 16
#define NUM_SEQ  2048
#define NUM_FEAT 1024
#define HEAD_DIM 64
#define NUM_B    4

typedef __attribute__((ext_vector_type(16))) __bf16    v16bf;
typedef __attribute__((ext_vector_type(8)))  float     v8f;
typedef __attribute__((ext_vector_type(4)))  uint32_t  u32x4;
typedef __attribute__((ext_vector_type(8)))  uint32_t  u32x8;

union Frag {
    v16bf bf;
    uint4 q[2];
};

__device__ __forceinline__ v8f zero8() {
    v8f z;
#pragma unroll
    for (int i = 0; i < 8; ++i) z[i] = 0.0f;
    return z;
}

__device__ __forceinline__ uint16_t f2bf(float f) {
    uint32_t u = __float_as_uint(f);
    u += 0x7FFF + ((u >> 16) & 1);   // round-to-nearest-even
    return (uint16_t)(u >> 16);
}

// 16x32 bf16 A-style fragment load (row-major source, contraction over last
// dim). ISA 16-bit A layout: lane L holds row m=L%16; lanes 0-15 own K
// offsets {0..7, 16..23}, lanes 16-31 own {8..15, 24..31}. Each half is 8
// contiguous bf16 = one 16-byte load (global_load_b128 / ds_load_b128).
__device__ __forceinline__ Frag load_frag(const uint16_t* base, int row0, int k0, int ld) {
    int lane = threadIdx.x & 31;
    int m  = lane & 15;
    int kb = (lane < 16) ? 0 : 8;
    const uint16_t* p = base + (size_t)(row0 + m) * ld + k0 + kb;
    Frag f;
    f.q[0] = *(const uint4*)(p);
    f.q[1] = *(const uint4*)(p + 16);
    return f;
}

__device__ __forceinline__ v8f wmma_bf16(const Frag& a, const Frag& b, v8f c) {
    return __builtin_amdgcn_wmma_f32_16x16x32_bf16(
        false, a.bf, false, b.bf, (short)0, c, false, false);
}

// ---------------------------------------------------------------------------
// Tensor Data Mover: 2D tile load Global -> LDS (bf16 elements).
// Builds D# group0 (count=1, lds_addr, global_addr, type=2) and group1
// (data_size=2B, tensor dims, tile dims, dim0 stride) per CDNA5 ISA §8 and
// issues tensor_load_to_lds with groups 2/3 omitted (NULL) -> up-to-2D tensor.
// Tracked with TENSORcnt. All operands workgroup-uniform -> SGPRs.
// tile_d1 == 0 means 1D (tile_dim1 unused).
// ---------------------------------------------------------------------------
__device__ __forceinline__ void tdm_load_2d(uint32_t lds_off, const void* gaddr,
                                            uint32_t tile_d0, uint32_t tile_d1,
                                            uint64_t stride0,
                                            uint64_t tensor_d0, uint64_t tensor_d1) {
    uint64_t ga = (uint64_t)(uintptr_t)gaddr;
    u32x4 g0;
    g0[0] = 1u;                                              // count=1 valid user D#
    g0[1] = lds_off;                                         // lds_addr (bytes)
    g0[2] = (uint32_t)ga;                                    // global_addr[31:0]
    g0[3] = (uint32_t)((ga >> 32) & 0x1FFFFFFu) | (2u << 30);// addr[56:32] | type=2
    u32x8 g1;
    g1[0] = 1u << 16;                                        // data_size=1 -> 2 bytes
    g1[1] = (uint32_t)((tensor_d0 & 0xFFFFu) << 16);         // tensor_dim0[15:0]
    g1[2] = (uint32_t)((tensor_d0 >> 16) & 0xFFFFu)          // tensor_dim0[31:16]
          | (uint32_t)((tensor_d1 & 0xFFFFu) << 16);         // tensor_dim1[15:0]
    g1[3] = (uint32_t)((tensor_d1 >> 16) & 0xFFFFu)          // tensor_dim1[31:16]
          | (tile_d0 << 16);                                 // tile_dim0
    g1[4] = tile_d1;                                         // tile_dim1 (tile_dim2=0)
    g1[5] = (uint32_t)(stride0 & 0xFFFFFFFFu);               // dim0_stride[31:0]
    g1[6] = (uint32_t)((stride0 >> 32) & 0xFFFFu);           // dim0_stride[47:32]
    g1[7] = 0u;
    asm volatile("tensor_load_to_lds %0, %1" :: "s"(g0), "s"(g1) : "memory");
}

// ---------------------------------------------------------------------------
// fp32 -> bf16 conversion
// ---------------------------------------------------------------------------
__global__ void cvt_f32_bf16(const float* __restrict__ in, uint16_t* __restrict__ out, int n) {
    int i = blockIdx.x * blockDim.x + threadIdx.x;
    if (i < n) out[i] = f2bf(in[i]);
}

// ---------------------------------------------------------------------------
// bf16 GEMM: C = A @ W^T   (A: MxK row-major, W: NxK row-major)
// Block = 128 threads (4 waves); wave w computes a 16x64 strip: rows
// [blockY*64+w*16,+16), cols [blockX*64,+64). A fragment reused across 4
// column tiles -> 2.5 B128 loads per WMMA.
// MODE 0: RoPE epilogue, store bf16 (b,h,s,hd)       (Q / K projections)
// MODE 2: store bf16 transposed (b,h,hd,s)           (V projection)
// MODE 3: store fp32 (M,N)                           (output projection)
// ---------------------------------------------------------------------------
template <int MODE>
__global__ void gemm_bf16_kernel(const uint16_t* __restrict__ A,
                                 const uint16_t* __restrict__ W,
                                 uint16_t* __restrict__ outb,
                                 float* __restrict__ outf,
                                 int M, int N, int K) {
    int wave = threadIdx.x >> 5;
    int lane = threadIdx.x & 31;
    int row0    = blockIdx.y * 64 + wave * 16;
    int colbase = blockIdx.x * 64;

    v8f acc[4];
#pragma unroll
    for (int ct = 0; ct < 4; ++ct) acc[ct] = zero8();

    for (int k0 = 0; k0 < K; k0 += 32) {
        Frag a = load_frag(A, row0, k0, K);
#pragma unroll
        for (int ct = 0; ct < 4; ++ct) {
            Frag b = load_frag(W, colbase + ct * 16, k0, K);
            acc[ct] = wmma_bf16(a, b, acc[ct]);
        }
    }

    int half = lane >> 4;
    int n    = lane & 15;

#pragma unroll
    for (int ct = 0; ct < 4; ++ct) {
        int col = colbase + ct * 16 + n;
#pragma unroll
        for (int r = 0; r < 8; ++r) {
            int m   = r + 8 * half;
            int row = row0 + m;
            float val = acc[ct][r];
            if (MODE == 0) {
                // RoPE: pairs (2i,2i+1) along features; partner is lane^1.
                float partner = __shfl_xor(val, 1, 32);
                int sidx = row & (NUM_SEQ - 1);
                int bb   = row >> 11;
                int head = col >> 6;
                int c    = col & 63;
                float theta = __powf(10000.0f, -(float)(c >> 1) * (1.0f / 31.0f));
                float sn, cs;
                __sincosf((float)sidx * theta, &sn, &cs);
                float rot = (c & 1) ? partner : -partner;
                float o   = val * cs + rot * sn;
                outb[((size_t)(bb * NUM_HEAD + head) * NUM_SEQ + sidx) * HEAD_DIM + c] = f2bf(o);
            } else if (MODE == 2) {
                int sidx = row & (NUM_SEQ - 1);
                int bb   = row >> 11;
                int head = col >> 6;
                int c    = col & 63;
                outb[((size_t)(bb * NUM_HEAD + head) * HEAD_DIM + c) * NUM_SEQ + sidx] = f2bf(val);
            } else {
                outf[(size_t)row * N + col] = val;
            }
        }
    }
}

// ---------------------------------------------------------------------------
// Retention attention, TDM-fed & double-buffered:
// workgroup = (64 q-rows, head, batch); 4 waves, wave w owns rows [q0,q0+16).
// Wave 0 issues tensor_load_to_lds for the NEXT key tile's K (64x64,
// contiguous) and V^T (64 feat-rows x 64 t, stride 2048) while all waves run
// WMMAs on the current LDS tiles; sync = s_wait_tensorcnt (wave0) + barrier.
// Decay mask factored exp((i-j)d) = exp(i*d)*exp(-j*d)  (|d|<=0.0317, |i*d|<=65
// -> no under/overflow). Scores round-trip LDS (bf16) to reshape C-layout ->
// A-fragment layout. Epilogue: LayerNorm over hd=64 + SiLU -> bf16 y.
// ---------------------------------------------------------------------------
__global__ void retention_attn_kernel(const uint16_t* __restrict__ q,
                                      const uint16_t* __restrict__ k,
                                      const uint16_t* __restrict__ vT,
                                      const float* __restrict__ lnw,
                                      const float* __restrict__ lnb,
                                      uint16_t* __restrict__ y) {
    __shared__ uint16_t kbuf[2][64 * 64];   // K tiles  (j_local x hd)
    __shared__ uint16_t vbuf[2][64 * 64];   // V^T tiles (c x t_local)
    __shared__ uint16_t sbuf[4][16 * 64];   // per-wave masked score tile

    int wave = threadIdx.x >> 5;
    int lane = threadIdx.x & 31;
    int qt = blockIdx.x;
    int h  = blockIdx.y;
    int b  = blockIdx.z;
    int q0 = qt * 64 + wave * 16;

    const uint16_t* qh = q  + (size_t)(b * NUM_HEAD + h) * NUM_SEQ * HEAD_DIM;
    const uint16_t* kh = k  + (size_t)(b * NUM_HEAD + h) * NUM_SEQ * HEAD_DIM;
    const uint16_t* vh = vT + (size_t)(b * NUM_HEAD + h) * HEAD_DIM * NUM_SEQ;

    float decay = logf(1.0f - exp2f(-5.0f - (float)h));
    int half = lane >> 4;
    int n    = lane & 15;

    Frag qf0 = load_frag(qh, q0, 0, HEAD_DIM);
    Frag qf1 = load_frag(qh, q0, 32, HEAD_DIM);

    float rowfac[8];
#pragma unroll
    for (int t = 0; t < 8; ++t)
        rowfac[t] = __expf((float)(q0 + t + 8 * half) * decay);

    v8f o[4];
#pragma unroll
    for (int ct = 0; ct < 4; ++ct) o[ct] = zero8();

    // Prefetch tile 0 (wave 0 issues; TDM ignores EXEC, SGPR descriptors).
    if (wave == 0) {
        tdm_load_2d((uint32_t)(uintptr_t)&kbuf[0][0], kh, 4096u, 0u,
                    4096ull, 4096ull, 1ull);
        tdm_load_2d((uint32_t)(uintptr_t)&vbuf[0][0], vh, 64u, 64u,
                    (uint64_t)NUM_SEQ, (uint64_t)NUM_SEQ, 64ull);
    }

    for (int jt = 0; jt <= qt; ++jt) {
        int j0  = jt * 64;
        int cur = jt & 1;

        if (wave == 0) {
            if (jt + 1 <= qt) {
                int nj0 = (jt + 1) * 64;
                int nb  = (jt + 1) & 1;
                tdm_load_2d((uint32_t)(uintptr_t)&kbuf[nb][0], kh + (size_t)nj0 * HEAD_DIM,
                            4096u, 0u, 4096ull, 4096ull, 1ull);
                tdm_load_2d((uint32_t)(uintptr_t)&vbuf[nb][0], vh + nj0,
                            64u, 64u, (uint64_t)NUM_SEQ, (uint64_t)NUM_SEQ, 64ull);
                __builtin_amdgcn_s_wait_tensorcnt(2);   // current tile landed
            } else {
                __builtin_amdgcn_s_wait_tensorcnt(0);
            }
        }
        __syncthreads();   // release all waves onto the current LDS tiles

        const uint16_t* kcur = &kbuf[cur][0];
        const uint16_t* vcur = &vbuf[cur][0];

        // ---- scores: q(16x64) @ k^T(64x64), K frags from LDS ----
        v8f r[4];
#pragma unroll
        for (int sub = 0; sub < 4; ++sub) {
            Frag kf0 = load_frag(kcur, sub * 16, 0, 64);
            Frag kf1 = load_frag(kcur, sub * 16, 32, 64);
            v8f rr = zero8();
            rr = wmma_bf16(qf0, kf0, rr);
            rr = wmma_bf16(qf1, kf1, rr);
            r[sub] = rr;
        }

        // ---- factored causal decay mask -> bf16 scores into LDS ----
#pragma unroll
        for (int sub = 0; sub < 4; ++sub) {
            int j = j0 + sub * 16 + n;
            float colfac = __expf(-(float)j * decay);
#pragma unroll
            for (int t = 0; t < 8; ++t) {
                int i = q0 + t + 8 * half;
                float mval = (i >= j) ? rowfac[t] * colfac : 0.0f;
                sbuf[wave][(t + 8 * half) * 64 + sub * 16 + n] = f2bf(r[sub][t] * mval);
            }
        }

        // ---- o += r(16x64) @ v(64x64); score + V frags from LDS ----
        Frag rf0 = load_frag(&sbuf[wave][0], 0, 0, 64);
        Frag rf1 = load_frag(&sbuf[wave][0], 0, 32, 64);
#pragma unroll
        for (int ct = 0; ct < 4; ++ct) {
            Frag vf0 = load_frag(vcur, ct * 16, 0, 64);
            Frag vf1 = load_frag(vcur, ct * 16, 32, 64);
            o[ct] = wmma_bf16(rf0, vf0, o[ct]);
            o[ct] = wmma_bf16(rf1, vf1, o[ct]);
        }
        __syncthreads();   // all reads of buf[cur] done before it is re-filled
    }

    // ---- LayerNorm over hd=64 + SiLU; rows live in 16-lane halves ----
#pragma unroll
    for (int t = 0; t < 8; ++t) {
        float s1 = 0.0f, s2 = 0.0f;
#pragma unroll
        for (int ct = 0; ct < 4; ++ct) {
            float v = o[ct][t];
            s1 += v;
            s2 += v * v;
        }
#pragma unroll
        for (int m = 1; m < 16; m <<= 1) {
            s1 += __shfl_xor(s1, m, 32);
            s2 += __shfl_xor(s2, m, 32);
        }
        float mean = s1 * (1.0f / 64.0f);
        float var  = s2 * (1.0f / 64.0f) - mean * mean;
        float rstd = rsqrtf(var + 1e-5f);
        int sidx = q0 + t + 8 * half;
#pragma unroll
        for (int ct = 0; ct < 4; ++ct) {
            int c = ct * 16 + n;
            float yv = (o[ct][t] - mean) * rstd * lnw[c] + lnb[c];
            yv = yv * (1.0f / (1.0f + __expf(-yv)));   // SiLU
            y[((size_t)b * NUM_SEQ + sidx) * NUM_FEAT + h * HEAD_DIM + c] = f2bf(yv);
        }
    }
}

// ---------------------------------------------------------------------------
// Host-side launcher
// ---------------------------------------------------------------------------
extern "C" void kernel_launch(void* const* d_in, const int* in_sizes, int n_in,
                              void* d_out, int out_size, void* d_ws, size_t ws_size,
                              hipStream_t stream) {
    const float* x   = (const float*)d_in[0];
    const float* Wq  = (const float*)d_in[1];
    const float* Wk  = (const float*)d_in[2];
    const float* Wv  = (const float*)d_in[3];
    const float* Wo  = (const float*)d_in[4];
    const float* lnw = (const float*)d_in[5];
    const float* lnb = (const float*)d_in[6];
    float* out = (float*)d_out;

    const int M = NUM_B * NUM_SEQ;                     // 8192
    const int D = NUM_FEAT;                            // 1024
    const size_t nX = (size_t)M * D;                   // 8388608
    const size_t nW = (size_t)D * D;                   // 1048576

    uint16_t* xb  = (uint16_t*)d_ws;
    uint16_t* wqb = xb  + nX;
    uint16_t* wkb = wqb + nW;
    uint16_t* wvb = wkb + nW;
    uint16_t* wob = wvb + nW;
    uint16_t* qb  = wob + nW;
    uint16_t* kb  = qb  + nX;
    uint16_t* vtb = kb  + nX;
    uint16_t* yb  = vtb + nX;
    (void)ws_size; (void)in_sizes; (void)n_in; (void)out_size;

    // 1) fp32 -> bf16 conversions
    cvt_f32_bf16<<<(int)((nX + 255) / 256), 256, 0, stream>>>(x,  xb,  (int)nX);
    cvt_f32_bf16<<<(int)((nW + 255) / 256), 256, 0, stream>>>(Wq, wqb, (int)nW);
    cvt_f32_bf16<<<(int)((nW + 255) / 256), 256, 0, stream>>>(Wk, wkb, (int)nW);
    cvt_f32_bf16<<<(int)((nW + 255) / 256), 256, 0, stream>>>(Wv, wvb, (int)nW);
    cvt_f32_bf16<<<(int)((nW + 255) / 256), 256, 0, stream>>>(Wo, wob, (int)nW);

    // 2) projections (WMMA GEMM + fused RoPE / V-transpose epilogues)
    dim3 ggrid(D / 64, M / 64);
    gemm_bf16_kernel<0><<<ggrid, 128, 0, stream>>>(xb, wqb, qb,  nullptr, M, D, D);
    gemm_bf16_kernel<0><<<ggrid, 128, 0, stream>>>(xb, wkb, kb,  nullptr, M, D, D);
    gemm_bf16_kernel<2><<<ggrid, 128, 0, stream>>>(xb, wvb, vtb, nullptr, M, D, D);

    // 3) TDM-fed streamed retention attention + LayerNorm + SiLU
    dim3 agrid(NUM_SEQ / 64, NUM_HEAD, NUM_B);
    retention_attn_kernel<<<agrid, 128, 0, stream>>>(qb, kb, vtb, lnw, lnb, yb);

    // 4) output projection: out = silu(y) @ Wo^T   (fp32 result)
    gemm_bf16_kernel<3><<<ggrid, 128, 0, stream>>>(yb, wob, nullptr, out, M, D, D);
}
